// DPCLoss_2267742733130
// MI455X (gfx1250) — matrix operand order repaired and checked by reading, not compiled, not run on hardware
//
#include <hip/hip_runtime.h>
#include <hip/hip_bf16.h>

#define M_TOT 4096
#define C_DIM 256
#define JT_PER_BLOCK 64      // j-tiles (16 cols each) handled per workgroup
#define ROW_Q 33             // 528 bytes per LDS panel row (512 data + 16 pad) in 16B units
#define PANEL_Q (16 * ROW_Q) // one 16-row panel

typedef __attribute__((ext_vector_type(16))) __bf16       v16bf;
typedef __attribute__((ext_vector_type(8)))  float        v8f;
typedef __attribute__((ext_vector_type(4)))  unsigned int u32x4;
typedef __attribute__((ext_vector_type(8)))  int          i32x8;
typedef __attribute__((ext_vector_type(4)))  int          i32x4;

struct alignas(16) Q4 { unsigned int x, y, z, w; };
union FragBF { Q4 q[2]; v16bf v; };

#ifndef USE_TDM
#if __has_builtin(__builtin_amdgcn_tensor_load_to_lds)
#define USE_TDM 1
#else
#define USE_TDM 0
#endif
#endif

static __device__ inline v8f wmma_bf16(v16bf a, v16bf b, v8f c) {
    // D = A(16x32 bf16) * B(32x16 bf16) + C(16x16 f32)
    return __builtin_amdgcn_wmma_f32_16x16x32_bf16(false, a, false, b, (short)0, c, false, false);
}

// fp32 -> bf16 hi/lo split (round-to-nearest-even both halves)
static __device__ inline void bsplit(float x, unsigned short& hi, unsigned short& lo) {
    unsigned u = __float_as_uint(x);
    unsigned r = u + 0x7FFFu + ((u >> 16) & 1u);
    unsigned short h = (unsigned short)(r >> 16);
    float hf = __uint_as_float(((unsigned)h) << 16);
    float l = x - hf;
    unsigned v = __float_as_uint(l);
    unsigned r2 = v + 0x7FFFu + ((v >> 16) & 1u);
    hi = h;
    lo = (unsigned short)(r2 >> 16);
}

// ---------------------------------------------------------------------------
// Kernel 1: (B,N,C,H,W) fp32 -> row-major (M,C) bf16 hi/lo for pred and gt.
// out index t = m*256 + c ; src index = bn*4096 + c*16 + hw  (bn=m>>4, hw=m&15)
// ---------------------------------------------------------------------------
__global__ __launch_bounds__(256) void convert_split_kernel(
    const float* __restrict__ pred, const float* __restrict__ gt,
    unsigned short* __restrict__ Ph, unsigned short* __restrict__ Pl,
    unsigned short* __restrict__ Gh, unsigned short* __restrict__ Gl)
{
    int t = blockIdx.x * 256 + threadIdx.x;        // 0 .. M*C-1
    int c = t & (C_DIM - 1);
    int mrow = t >> 8;
    int bn = mrow >> 4;
    int hw = mrow & 15;
    int src = bn * 4096 + c * 16 + hw;
    bsplit(pred[src], Ph[t], Pl[t]);
    bsplit(gt[src],   Gh[t], Gl[t]);
}

// ---------------------------------------------------------------------------
// Kernel 2: exact fp32 diagonal  diag[i] = dot(P_i, G_i)
// one wave per row, lane-strided over C, shuffle reduce
// ---------------------------------------------------------------------------
__global__ __launch_bounds__(256) void diag_kernel(
    const float* __restrict__ pred, const float* __restrict__ gt,
    float* __restrict__ diag)
{
    int wave = threadIdx.x >> 5, lane = threadIdx.x & 31;
    int row = blockIdx.x * 8 + wave;
    int bn = row >> 4, hw = row & 15;
    const float* p = pred + bn * 4096 + hw;
    const float* g = gt   + bn * 4096 + hw;
    float acc = 0.f;
    for (int c = lane; c < C_DIM; c += 32)
        acc += p[c * 16] * g[c * 16];
    #pragma unroll
    for (int mask = 16; mask >= 1; mask >>= 1)
        acc += __shfl_xor(acc, mask, 32);
    if (lane == 0) diag[row] = acc;
}

// ---------------------------------------------------------------------------
// TDM panel load: 2D tile, 16 rows x 256 bf16 (512B) per row, data_size=2B,
// LDS padding: 16B after every 512B row -> 528B pitch (pad_interval code 6 =
// 128 dwords, pad_amount code 3 = 4 dwords).  Descriptor per ISA ch.8.
// ---------------------------------------------------------------------------
#if USE_TDM
static __device__ inline void tdm_panel(const unsigned short* gsrc, int rowStart,
                                        unsigned ldsByteOff)
{
    unsigned long long ga = (unsigned long long)(uintptr_t)(gsrc + (size_t)rowStart * C_DIM);
    u32x4 g0;
    g0[0] = 1u;                                            // count=1, user mode
    g0[1] = ldsByteOff;                                    // lds_addr
    g0[2] = (unsigned)ga;                                  // global_addr[31:0]
    g0[3] = ((unsigned)(ga >> 32) & 0x1FFFFFFu) | (2u << 30); // addr[56:32] | type=2
    i32x8 g1;
    g1[0] = (int)((1u << 16) | (1u << 20) | (6u << 22) | (3u << 25)); // 2B elems, pad on
    g1[1] = (int)(((unsigned)C_DIM) << 16);                // tensor_dim0 lo16
    g1[2] = (int)(((unsigned)M_TOT) << 16);                // dim0 hi16=0 | tensor_dim1 lo16
    g1[3] = (int)(((unsigned)C_DIM) << 16);                // dim1 hi16=0 | tile_dim0=256
    g1[4] = 16;                                            // tile_dim1=16, tile_dim2=0
    g1[5] = C_DIM;                                         // tensor_dim0_stride lo32
    g1[6] = 0;
    g1[7] = 0;
    i32x4 z4 = {0, 0, 0, 0};
    i32x8 z8 = {0, 0, 0, 0, 0, 0, 0, 0};
    __builtin_amdgcn_tensor_load_to_lds(g0, g1, z4, z4, z8, 0);
}
#else
static __device__ inline void stage_fallback(Q4* dstPanel, const unsigned short* gsrc,
                                             int rowStart, int tid)
{
    const Q4* src = reinterpret_cast<const Q4*>(gsrc);
    #pragma unroll
    for (int k = 0; k < 4; ++k) {
        int id = tid + k * 128;       // 512 16B-chunks per panel
        int row = id >> 5;
        int cir = id & 31;
        dstPanel[row * ROW_Q + cir] = src[(size_t)(rowStart + row) * 32 + cir];
    }
}
#endif

// ---------------------------------------------------------------------------
// Kernel 3: fused GEMM + online softmax/argmax.
// grid = (64 row-blocks of 64 rows, 4 column splits of 1024 cols), 128 thr.
// Each wave owns a 16-row A-tile (held in registers, hi+lo), loops over 64
// j-tiles whose B panels are TDM double-buffered in LDS.
// Compensated product:  Ph*Gh' + Ph*Gl' + Pl*Gh'  (3 bf16 WMMAs / K-step).
// ---------------------------------------------------------------------------
__global__ __launch_bounds__(128) void dpc_main(
    const unsigned short* __restrict__ Ph, const unsigned short* __restrict__ Pl,
    const unsigned short* __restrict__ Gh, const unsigned short* __restrict__ Gl,
    float* __restrict__ pm, float* __restrict__ ps,
    float* __restrict__ pbv, int* __restrict__ pbj)
{
    __shared__ Q4 lds[2][2][PANEL_Q];   // [buffer][hi/lo][panel]

    const int tid  = threadIdx.x;
    const int lane = tid & 31;
    const int wave = tid >> 5;
    const int half = lane >> 4;          // 0: lanes 0-15, 1: lanes 16-31
    const int r0    = blockIdx.x * 64 + wave * 16;
    const int jbase = blockIdx.y * JT_PER_BLOCK;
    const int rowA  = r0 + (lane & 15);

    // A fragments for all 8 K-steps, hi and lo (ISA 16-bit A layout:
    // lane chunk1 = K [k0 + half*8, +8), chunk2 = +16 elements)
    v16bf ah[8], al[8];
    #pragma unroll
    for (int ks = 0; ks < 8; ++ks) {
        const Q4* ph = reinterpret_cast<const Q4*>(Ph + (size_t)rowA * C_DIM + ks * 32 + half * 8);
        const Q4* pl = reinterpret_cast<const Q4*>(Pl + (size_t)rowA * C_DIM + ks * 32 + half * 8);
        FragBF fa, fb;
        fa.q[0] = ph[0]; fa.q[1] = ph[2];
        fb.q[0] = pl[0]; fb.q[1] = pl[2];
        ah[ks] = fa.v; al[ks] = fb.v;
    }

    float m[8], s[8], bv[8]; int bj[8];
    #pragma unroll
    for (int r = 0; r < 8; ++r) { m[r] = -3.0e38f; s[r] = 0.f; bv[r] = -3.0e38f; bj[r] = 0; }

    // prologue: stage first panel into buffer 0
#if USE_TDM
    if (wave == 0) {
        tdm_panel(Gh, jbase * 16, (unsigned)(uintptr_t)&lds[0][0][0]);
        tdm_panel(Gl, jbase * 16, (unsigned)(uintptr_t)&lds[0][1][0]);
    }
    __builtin_amdgcn_s_wait_tensorcnt(0);
#else
    stage_fallback(&lds[0][0][0], Gh, jbase * 16, tid);
    stage_fallback(&lds[0][1][0], Gl, jbase * 16, tid);
#endif
    __syncthreads();

    #pragma unroll 1
    for (int jt = 0; jt < JT_PER_BLOCK; ++jt) {
        const int buf = jt & 1;
        if (jt + 1 < JT_PER_BLOCK) {
#if USE_TDM
            if (wave == 0) {
                tdm_panel(Gh, (jbase + jt + 1) * 16, (unsigned)(uintptr_t)&lds[buf ^ 1][0][0]);
                tdm_panel(Gl, (jbase + jt + 1) * 16, (unsigned)(uintptr_t)&lds[buf ^ 1][1][0]);
            }
#else
            stage_fallback(&lds[buf ^ 1][0][0], Gh, (jbase + jt + 1) * 16, tid);
            stage_fallback(&lds[buf ^ 1][1][0], Gl, (jbase + jt + 1) * 16, tid);
#endif
        }

        v8f acc = {0.f, 0.f, 0.f, 0.f, 0.f, 0.f, 0.f, 0.f};
        const Q4* panelH = &lds[buf][0][0];
        const Q4* panelL = &lds[buf][1][0];
        #pragma unroll
        for (int ks = 0; ks < 8; ++ks) {
            // B layout: lane col = lane&15, K chunks = [k0 + half*8, +8) and [k0 + 16 + half*8, +8)
            int bidx = (lane & 15) * ROW_Q + ks * 4 + half;
            FragBF fh, fl;
            fh.q[0] = panelH[bidx]; fh.q[1] = panelH[bidx + 2];
            fl.q[0] = panelL[bidx]; fl.q[1] = panelL[bidx + 2];
            acc = wmma_bf16(ah[ks], fh.v, acc);   // hi*hi
            acc = wmma_bf16(ah[ks], fl.v, acc);   // hi*lo
            acc = wmma_bf16(al[ks], fh.v, acc);   // lo*hi
        }

        // online softmax / argmax update; lane owns col, rows r + 8*half
        const int col = (jbase + jt) * 16 + (lane & 15);
        #pragma unroll
        for (int r = 0; r < 8; ++r) {
            float v = acc[r];
            if (v > bv[r]) { bv[r] = v; bj[r] = col; }
            if (v <= m[r]) {
                s[r] += __expf(v - m[r]);
            } else {
                s[r] = s[r] * __expf(m[r] - v) + 1.0f;
                m[r] = v;
            }
        }

#if USE_TDM
        __builtin_amdgcn_s_wait_tensorcnt(0);
#endif
        __syncthreads();
    }

    // merge across the 16 lanes of each half (bits 0..3 of lane id)
    #pragma unroll
    for (int mask = 1; mask <= 8; mask <<= 1) {
        #pragma unroll
        for (int r = 0; r < 8; ++r) {
            float om = __shfl_xor(m[r], mask, 32);
            float os = __shfl_xor(s[r], mask, 32);
            float nm = fmaxf(m[r], om);
            s[r] = s[r] * __expf(m[r] - nm) + os * __expf(om - nm);
            m[r] = nm;
            float obv = __shfl_xor(bv[r], mask, 32);
            int   obj = __shfl_xor(bj[r], mask, 32);
            if (obv > bv[r] || (obv == bv[r] && obj < bj[r])) { bv[r] = obv; bj[r] = obj; }
        }
    }

    if ((lane & 15) == 0) {
        const int slot = blockIdx.y;
        const int rbase = r0 + half * 8;
        #pragma unroll
        for (int r = 0; r < 8; ++r) {
            int row = rbase + r;
            pm [slot * M_TOT + row] = m[r];
            ps [slot * M_TOT + row] = s[r];
            pbv[slot * M_TOT + row] = bv[r];
            pbj[slot * M_TOT + row] = bj[r];
        }
    }
}

// ---------------------------------------------------------------------------
// Kernel 4: merge 4 column-split partials per row, reduce loss + accuracy.
// ---------------------------------------------------------------------------
__global__ __launch_bounds__(256) void finalize_kernel(
    const float* __restrict__ pm, const float* __restrict__ ps,
    const float* __restrict__ pbv, const int* __restrict__ pbj,
    const float* __restrict__ diag, float* __restrict__ out)
{
    __shared__ float sl[256];
    __shared__ float sc[256];
    int tid = threadIdx.x;
    float lossAcc = 0.f, corr = 0.f;
    for (int row = tid; row < M_TOT; row += 256) {
        float mm = -3.0e38f;
        #pragma unroll
        for (int k = 0; k < 4; ++k) mm = fmaxf(mm, pm[k * M_TOT + row]);
        float ssum = 0.f;
        #pragma unroll
        for (int k = 0; k < 4; ++k)
            ssum += ps[k * M_TOT + row] * __expf(pm[k * M_TOT + row] - mm);
        float lse = mm + __logf(ssum);
        float bestv = -3.0e38f; int bestj = 0x7fffffff;
        #pragma unroll
        for (int k = 0; k < 4; ++k) {
            float v = pbv[k * M_TOT + row];
            int   j = pbj[k * M_TOT + row];
            if (v > bestv || (v == bestv && j < bestj)) { bestv = v; bestj = j; }
        }
        lossAcc += lse - diag[row];
        corr += (bestj == row) ? 1.f : 0.f;
    }
    sl[tid] = lossAcc; sc[tid] = corr;
    __syncthreads();
    for (int off = 128; off > 0; off >>= 1) {
        if (tid < off) { sl[tid] += sl[tid + off]; sc[tid] += sc[tid + off]; }
        __syncthreads();
    }
    if (tid == 0) {
        out[0] = sl[0] / (float)M_TOT;          // loss
        out[1] = 100.0f * sc[0] / (float)M_TOT; // accuracy
    }
}

// ---------------------------------------------------------------------------
extern "C" void kernel_launch(void* const* d_in, const int* in_sizes, int n_in,
                              void* d_out, int out_size, void* d_ws, size_t ws_size,
                              hipStream_t stream)
{
    (void)in_sizes; (void)n_in; (void)out_size; (void)ws_size;
    const float* pred = (const float*)d_in[0];
    const float* gt   = (const float*)d_in[1];
    float* out = (float*)d_out;

    char* w = (char*)d_ws;
    const size_t MC = (size_t)M_TOT * C_DIM;   // 1M elements
    unsigned short* Ph = (unsigned short*)w;
    unsigned short* Pl = Ph + MC;
    unsigned short* Gh = Pl + MC;
    unsigned short* Gl = Gh + MC;
    float* diag = (float*)(w + 8 * MC);        // after 4 * 2MB of bf16
    float* pm   = diag + M_TOT;
    float* ps   = pm + 4 * M_TOT;
    float* pbv  = ps + 4 * M_TOT;
    int*   pbj  = (int*)(pbv + 4 * M_TOT);

    convert_split_kernel<<<(M_TOT * C_DIM) / 256, 256, 0, stream>>>(pred, gt, Ph, Pl, Gh, Gl);
    diag_kernel<<<M_TOT / 8, 256, 0, stream>>>(pred, gt, diag);
    dpc_main<<<dim3(64, 4), 128, 0, stream>>>(Ph, Pl, Gh, Gl, pm, ps, pbv, pbj);
    finalize_kernel<<<1, 256, 0, stream>>>(pm, ps, pbv, pbj, diag, out);
}
